// STMambaBlock_57543971832010
// MI455X (gfx1250) — matrix operand order repaired and checked by reading, not compiled, not run on hardware
//
#include <hip/hip_runtime.h>
#include <hip/hip_bf16.h>
#include <math.h>
#include <stdint.h>

typedef __attribute__((ext_vector_type(16))) _Float16 v16h;
typedef __attribute__((ext_vector_type(8)))  _Float16 v8h;
typedef __attribute__((ext_vector_type(4)))  _Float16 v4h;
typedef __attribute__((ext_vector_type(8)))  float    v8f;
typedef __attribute__((ext_vector_type(4)))  unsigned int u32x4;
typedef __attribute__((ext_vector_type(8)))  int      i32x8;
typedef __attribute__((ext_vector_type(4)))  int      i32x4;

#if defined(__has_builtin)
#if __has_builtin(__builtin_amdgcn_tensor_load_to_lds)
#define HAVE_TDM 1
#endif
#endif
#ifndef HAVE_TDM
#define HAVE_TDM 0
#endif

#define NNODE   128
#define DM      128
#define HEADS   4
#define HD      32
#define DI      256      // D_INNER
#define DS      16       // D_STATE
#define DTR     8        // DT_RANK
#define BSZ     4
#define TLEN    32
#define BT      128      // BSZ*TLEN
#define ROWS    16384    // BT*NNODE
#define BN      512      // BSZ*NNODE

// ---------------- WMMA helpers (CDNA5 16x16x32 f16 -> f32) ----------------

__device__ __forceinline__ v8f zero8() {
  v8f z;
#pragma unroll
  for (int i = 0; i < 8; ++i) z[i] = 0.f;
  return z;
}

__device__ __forceinline__ v8f wmma16(v16h a, v16h b, v8f c) {
  return __builtin_amdgcn_wmma_f32_16x16x32_f16(false, a, false, b, (short)0, c,
                                                false, false);
}

// A-matrix 16x32 f16 layout: contiguous 16-half run per lane (k = hf*16+e),
// matching the B-fragment k order.
__device__ __forceinline__ v16h load_A16(const _Float16* p, int stride,
                                         int mbase, int kbase, int lane) {
  int hf = (lane >> 4) & 1, mr = lane & 15;
  const _Float16* row = p + (size_t)(mbase + mr) * stride + kbase + hf * 16;
  v16h a;
#pragma unroll
  for (int e = 0; e < 16; ++e) a[e] = row[e];
  return a;
}

// B operand pre-swizzled into register-fragment order: one 32B contiguous
// v16h per (fragment, lane) -> exactly 2x ds_load_b128 per WMMA operand.
__device__ __forceinline__ v16h load_frag(const _Float16* p, int frag,
                                          int lane) {
  return *(const v16h*)(p + (size_t)(frag * 32 + lane) * 16);
}

// B = srcT (source row-major N x K): contiguous 16-half run per lane.
__device__ __forceinline__ v16h load_BT16(const _Float16* p, int stride,
                                          int kbase, int nbase, int lane) {
  int hf = (lane >> 4) & 1, nc = lane & 15;
  const _Float16* row = p + (size_t)(nbase + nc) * stride + kbase + hf * 16;
  v16h b;
#pragma unroll
  for (int e = 0; e < 16; ++e) b[e] = row[e];
  return b;
}

// ---- Tensor Data Mover: async 2D f16 tile copy global -> LDS -------------
// D# per ISA 8.3/8.4: group0 = {count=1, lds_addr, global_addr, type=2},
// group1 = {data_size=2B, tensor_dim0/1, tile_dim0/1, tensor_dim0_stride}.
// Issue from one wave; drain with s_wait_tensorcnt before the barrier.
__device__ __forceinline__ void tdm_load_2d_f16(_Float16* lds_dst,
                                                const _Float16* gsrc,
                                                unsigned rows, unsigned cols,
                                                unsigned src_stride) {
#if HAVE_TDM
  unsigned lds_addr = (unsigned)(uintptr_t)lds_dst; // low 32b = LDS offset
  unsigned long long ga = (unsigned long long)(uintptr_t)gsrc;
  u32x4 g0;
  g0[0] = 1u;                                        // count=1, user D#
  g0[1] = lds_addr;                                  // lds_addr
  g0[2] = (unsigned)(ga & 0xFFFFFFFFull);            // global_addr[31:0]
  g0[3] = (unsigned)((ga >> 32) & 0x01FFFFFFull) | (2u << 30); // type=2
  i32x8 g1;
  g1[0] = (int)(1u << 16);                           // data_size = 2 bytes
  g1[1] = (int)((cols & 0xFFFFu) << 16);             // tensor_dim0 lo16
  g1[2] = (int)((cols >> 16) | ((rows & 0xFFFFu) << 16)); // dim0 hi | dim1 lo
  g1[3] = (int)((rows >> 16) | ((cols & 0xFFFFu) << 16)); // dim1 hi | tile_dim0
  g1[4] = (int)(rows & 0xFFFFu);                     // tile_dim1 (tile_dim2=0)
  g1[5] = (int)src_stride;                           // tensor_dim0_stride lo32
  g1[6] = 0;
  g1[7] = 0;
  i32x4 z4;
  z4[0] = z4[1] = z4[2] = z4[3] = 0;
#if __clang_major__ >= 23
  i32x8 z8;
#pragma unroll
  for (int i = 0; i < 8; ++i) z8[i] = 0;
  __builtin_amdgcn_tensor_load_to_lds(g0, g1, z4, z4, z8, 0);
#else
  __builtin_amdgcn_tensor_load_to_lds(g0, g1, z4, z4, 0);
#endif
#endif
}

__device__ __forceinline__ void tdm_drain() {
#if HAVE_TDM
  __builtin_amdgcn_s_wait_tensorcnt(0);
#endif
}

// ---- cooperative staging (256 threads) ----------------------------------

// f32 global -> f16 LDS row-major (vectorized: float4 in, v4h out)
__device__ __forceinline__ void stage_rows_f32(_Float16* dst,
                                               const float* __restrict__ src,
                                               int count, int tid) {
  for (int i = tid * 4; i < count; i += 256 * 4) {
    float4 v4 = *(const float4*)(src + i);
    v4h h4 = {(_Float16)v4.x, (_Float16)v4.y, (_Float16)v4.z, (_Float16)v4.w};
    *(v4h*)(dst + i) = h4;
  }
}

// f16 global -> f16 LDS row-major (16B chunks)
__device__ __forceinline__ void stage_rows_f16(_Float16* dst,
                                               const _Float16* __restrict__ src,
                                               int count, int tid) {
  for (int i = tid * 8; i < count; i += 256 * 8)
    *(v8h*)(dst + i) = *(const v8h*)(src + i);
}

// f32 row-major KxN weight -> B-fragment LDS layout:
// frag = kc*NT+nt, lane = hf*16+nc, elem e;  B[k][n], k = kc*32+hf*16+e.
template <int K, int N>
__device__ __forceinline__ void stage_Bfrag_f32(_Float16* dst,
                                                const float* __restrict__ src,
                                                int ldsrc, int colbase,
                                                int tid) {
  constexpr int NT = N / 16;
  for (int i = tid * 4; i < K * N; i += 256 * 4) {
    int k = i / N, n = i % N;
    float4 v4 = *(const float4*)(src + (size_t)k * ldsrc + colbase + n);
    int kc = k >> 5, kk = k & 31, hf = kk >> 4, e = kk & 15;
    float vv[4] = {v4.x, v4.y, v4.z, v4.w};
#pragma unroll
    for (int j = 0; j < 4; ++j) {
      int nn = n + j;
      dst[(size_t)(((kc * NT + (nn >> 4)) * 32 + hf * 16 + (nn & 15)) * 16 +
                   e)] = (_Float16)vv[j];
    }
  }
}

// f16 strided source -> B-fragment LDS layout (for v head slices)
template <int K, int N>
__device__ __forceinline__ void stage_Bfrag_f16(_Float16* dst,
                                                const _Float16* __restrict__ src,
                                                int ldsrc, int tid) {
  constexpr int NT = N / 16;
  for (int i = tid * 4; i < K * N; i += 256 * 4) {
    int k = i / N, n = i % N;
    v4h s4 = *(const v4h*)(src + (size_t)k * ldsrc + n);
    int kc = k >> 5, kk = k & 31, hf = kk >> 4, e = kk & 15;
#pragma unroll
    for (int j = 0; j < 4; ++j) {
      int nn = n + j;
      dst[(size_t)(((kc * NT + (nn >> 4)) * 32 + hf * 16 + (nn & 15)) * 16 +
                   e)] = s4[j];
    }
  }
}

// reductions across the 16 lanes sharing one C-layout row (wave32 halves)
__device__ __forceinline__ float hsum16(float v) {
  v += __shfl_xor(v, 1, 32);
  v += __shfl_xor(v, 2, 32);
  v += __shfl_xor(v, 4, 32);
  v += __shfl_xor(v, 8, 32);
  return v;
}
__device__ __forceinline__ float hmax16(float v) {
  v = fmaxf(v, __shfl_xor(v, 1, 32));
  v = fmaxf(v, __shfl_xor(v, 2, 32));
  v = fmaxf(v, __shfl_xor(v, 4, 32));
  v = fmaxf(v, __shfl_xor(v, 8, 32));
  return v;
}

// ---------------- 1a: q,k,v projections --------------------------------

__global__ void k_qkv(const float* __restrict__ x, const float* __restrict__ Wq,
                      const float* __restrict__ Wk, const float* __restrict__ Wv,
                      _Float16* __restrict__ qo, _Float16* __restrict__ ko,
                      _Float16* __restrict__ vo) {
  __shared__ __align__(32) _Float16 sA[128 * 128];
  __shared__ __align__(32) _Float16 sBf[128 * 128];
  int tid = threadIdx.x, lane = tid & 31, wid = tid >> 5;
  int hf = (lane >> 4) & 1, nc = lane & 15;
  int rowbase = blockIdx.x * 128, mbase = wid * 16;

  stage_rows_f32(sA, x + (size_t)rowbase * 128, 128 * 128, tid);

  const float* Ws[3] = {Wq, Wk, Wv};
  _Float16* Os[3] = {qo, ko, vo};

  for (int w = 0; w < 3; ++w) {
    __syncthreads();
    stage_Bfrag_f32<128, 128>(sBf, Ws[w], 128, 0, tid);
    if (w < 2) __builtin_prefetch(Ws[w + 1] + tid * 64, 0, 1); // warm L2
    __syncthreads();
    v8f acc[8];
#pragma unroll
    for (int nt = 0; nt < 8; ++nt) acc[nt] = zero8();
#pragma unroll
    for (int kc = 0; kc < 4; ++kc) {
      v16h a = load_A16(sA, 128, mbase, kc * 32, lane);
#pragma unroll
      for (int nt = 0; nt < 8; ++nt) {
        v16h b = load_frag(sBf, kc * 8 + nt, lane);
        acc[nt] = wmma16(a, b, acc[nt]);
      }
    }
    _Float16* O = Os[w];
#pragma unroll
    for (int nt = 0; nt < 8; ++nt)
#pragma unroll
      for (int r = 0; r < 8; ++r)
        O[(size_t)(rowbase + mbase + r + 8 * hf) * 128 + nt * 16 + nc] =
            (_Float16)acc[nt][r];
  }
}

// ---------------- 1b: masked attention per (graph, head) ----------------

__global__ void k_attn(const _Float16* __restrict__ q,
                       const _Float16* __restrict__ k,
                       const _Float16* __restrict__ v,
                       const float* __restrict__ adj, float* __restrict__ o) {
  __shared__ __align__(32) _Float16 sQ[128 * 32];
  __shared__ __align__(32) _Float16 sK[128 * 32];
  __shared__ __align__(32) _Float16 sVf[128 * 32];
  __shared__ __align__(32) _Float16 sAttn[128 * 128];
  int tid = threadIdx.x, lane = tid & 31, wid = tid >> 5;
  int hf = (lane >> 4) & 1, nc = lane & 15;
  int bt = blockIdx.x, head = blockIdx.y, mbase = wid * 16;

  // TDM: async 128x32 f16 tiles (row pitch 128) into LDS, issued by wave 0
  // while all waves stage the swizzled v fragments.
#if HAVE_TDM
  if (wid == 0 && lane == 0) {
    const _Float16* base = q + (size_t)(bt * 128) * 128 + head * 32;
    tdm_load_2d_f16(sQ, base, 128, 32, 128);
    tdm_load_2d_f16(sK, k + (size_t)(bt * 128) * 128 + head * 32, 128, 32, 128);
  }
#else
  for (int i = tid * 8; i < 128 * 32; i += 256 * 8) {
    int n = i >> 5, d = i & 31;
    size_t src = (size_t)(bt * 128 + n) * 128 + head * 32 + d;
    *(v8h*)(sQ + i) = *(const v8h*)(q + src);
    *(v8h*)(sK + i) = *(const v8h*)(k + src);
  }
#endif
  stage_Bfrag_f16<128, 32>(sVf, v + (size_t)(bt * 128) * 128 + head * 32, 128,
                           tid);
#if HAVE_TDM
  if (wid == 0) tdm_drain();
#endif
  __syncthreads();

  // scores = q . k^T / sqrt(HD); K dim = HD = 32 -> one WMMA per 16x16 tile
  v8f s[8];
  v16h aq = load_A16(sQ, 32, mbase, 0, lane);
#pragma unroll
  for (int mt = 0; mt < 8; ++mt) {
    v16h b = load_BT16(sK, 32, 0, mt * 16, lane);
    s[mt] = wmma16(aq, b, zero8());
  }

  const float scale = 0.17677669529663687f; // 1/sqrt(32)
#pragma unroll
  for (int mt = 0; mt < 8; ++mt)
#pragma unroll
    for (int r = 0; r < 8; ++r) {
      int n = mbase + r + 8 * hf, m = mt * 16 + nc;
      float vv = s[mt][r] * scale;
      if (adj[n * 128 + m] == 0.f) vv = -1e9f;
      s[mt][r] = vv;
    }

  // register softmax per row (row lives in 16 lanes x 8 tile-regs)
#pragma unroll
  for (int r = 0; r < 8; ++r) {
    float mx = -3.0e38f;
#pragma unroll
    for (int mt = 0; mt < 8; ++mt) mx = fmaxf(mx, s[mt][r]);
    mx = hmax16(mx);
    float sm = 0.f;
#pragma unroll
    for (int mt = 0; mt < 8; ++mt) {
      float e = __expf(s[mt][r] - mx);
      s[mt][r] = e;
      sm += e;
    }
    sm = hsum16(sm);
    float inv = 1.f / sm;
#pragma unroll
    for (int mt = 0; mt < 8; ++mt) s[mt][r] *= inv;
  }

#pragma unroll
  for (int mt = 0; mt < 8; ++mt)
#pragma unroll
    for (int r = 0; r < 8; ++r)
      sAttn[(size_t)(mbase + r + 8 * hf) * 128 + mt * 16 + nc] =
          (_Float16)s[mt][r];
  __syncthreads();

  // o = attn @ v_h   (K = 128 over nodes, N = 32 head dims)
#pragma unroll
  for (int nt = 0; nt < 2; ++nt) {
    v8f acc = zero8();
#pragma unroll
    for (int kc = 0; kc < 4; ++kc) {
      v16h a = load_A16(sAttn, 128, mbase, kc * 32, lane);
      v16h b = load_frag(sVf, kc * 2 + nt, lane);
      acc = wmma16(a, b, acc);
    }
#pragma unroll
    for (int r = 0; r < 8; ++r)
      o[(size_t)(bt * 128 + mbase + r + 8 * hf) * 128 + head * 32 + nt * 16 +
        nc] = acc[r];
  }
}

// ---------------- 1c: o@Wo + bo + residual + LayerNorm ------------------

__global__ void k_proj_ln(const float* __restrict__ of,
                          const float* __restrict__ Wo,
                          const float* __restrict__ bo,
                          const float* __restrict__ x,
                          const float* __restrict__ g,
                          const float* __restrict__ bb,
                          float* __restrict__ h2, _Float16* __restrict__ h2h) {
  __shared__ __align__(32) _Float16 sA[128 * 128];
  __shared__ __align__(32) _Float16 sBf[128 * 128];
  int tid = threadIdx.x, lane = tid & 31, wid = tid >> 5;
  int hf = (lane >> 4) & 1, nc = lane & 15;
  int rowbase = blockIdx.x * 128, mbase = wid * 16;

  stage_rows_f32(sA, of + (size_t)rowbase * 128, 128 * 128, tid);
  stage_Bfrag_f32<128, 128>(sBf, Wo, 128, 0, tid);
  __syncthreads();

  v8f acc[8];
#pragma unroll
  for (int nt = 0; nt < 8; ++nt) acc[nt] = zero8();
#pragma unroll
  for (int kc = 0; kc < 4; ++kc) {
    v16h a = load_A16(sA, 128, mbase, kc * 32, lane);
#pragma unroll
    for (int nt = 0; nt < 8; ++nt) {
      v16h b = load_frag(sBf, kc * 8 + nt, lane);
      acc[nt] = wmma16(a, b, acc[nt]);
    }
  }

  // epilogue: + bo + residual, then rowwise LayerNorm in registers
#pragma unroll
  for (int nt = 0; nt < 8; ++nt)
#pragma unroll
    for (int r = 0; r < 8; ++r) {
      int col = nt * 16 + nc;
      int row = rowbase + mbase + r + 8 * hf;
      acc[nt][r] += bo[col] + x[(size_t)row * 128 + col];
    }
#pragma unroll
  for (int r = 0; r < 8; ++r) {
    float s1 = 0.f, s2 = 0.f;
#pragma unroll
    for (int nt = 0; nt < 8; ++nt) {
      s1 += acc[nt][r];
      s2 += acc[nt][r] * acc[nt][r];
    }
    s1 = hsum16(s1);
    s2 = hsum16(s2);
    float mean = s1 * (1.f / 128.f);
    float var = s2 * (1.f / 128.f) - mean * mean;
    float rstd = rsqrtf(var + 1e-5f);
    int row = rowbase + mbase + r + 8 * hf;
#pragma unroll
    for (int nt = 0; nt < 8; ++nt) {
      int col = nt * 16 + nc;
      float vv = (acc[nt][r] - mean) * rstd * g[col] + bb[col];
      h2[(size_t)row * 128 + col] = vv;
      h2h[(size_t)row * 128 + col] = (_Float16)vv;
    }
  }
}

// ---------------- 2a: in_proj GEMM with (b,t,n)->(b,n,t) permuted store --

__global__ void k_inproj(const _Float16* __restrict__ h2h,
                         const float* __restrict__ in_proj,
                         _Float16* __restrict__ xz) {
  __shared__ __align__(32) _Float16 sA[128 * 128];
  __shared__ __align__(32) _Float16 sBf[128 * 128];
  int tid = threadIdx.x, lane = tid & 31, wid = tid >> 5;
  int hf = (lane >> 4) & 1, nc = lane & 15;
  int rowbase = blockIdx.x * 128, colbase = blockIdx.y * 128, mbase = wid * 16;

  // TDM: async contiguous 128x128 f16 tile into LDS (one wave issues),
  // while everyone swizzles the weight tile into B-fragment layout.
#if HAVE_TDM
  if (wid == 0 && lane == 0)
    tdm_load_2d_f16(sA, h2h + (size_t)rowbase * 128, 128, 128, 128);
#else
  stage_rows_f16(sA, h2h + (size_t)rowbase * 128, 128 * 128, tid);
#endif
  stage_Bfrag_f32<128, 128>(sBf, in_proj, 512, colbase, tid);
#if HAVE_TDM
  if (wid == 0) tdm_drain();
#endif
  __syncthreads();

  v8f acc[8];
#pragma unroll
  for (int nt = 0; nt < 8; ++nt) acc[nt] = zero8();
#pragma unroll
  for (int kc = 0; kc < 4; ++kc) {
    v16h a = load_A16(sA, 128, mbase, kc * 32, lane);
#pragma unroll
    for (int nt = 0; nt < 8; ++nt) {
      v16h b = load_frag(sBf, kc * 8 + nt, lane);
      acc[nt] = wmma16(a, b, acc[nt]);
    }
  }
#pragma unroll
  for (int nt = 0; nt < 8; ++nt)
#pragma unroll
    for (int r = 0; r < 8; ++r) {
      int gr = rowbase + mbase + r + 8 * hf;      // (b,t,n) order
      int b = gr >> 12, t = (gr >> 7) & 31, n = gr & 127;
      int ro = ((b << 7) + n) * 32 + t;           // (b,n,t) order
      xz[(size_t)ro * 512 + colbase + nt * 16 + nc] = (_Float16)acc[nt][r];
    }
}

// ---------------- 2b: causal depthwise conv + SiLU ----------------------

__global__ void k_conv(const _Float16* __restrict__ xz,
                       const float* __restrict__ conv_w,
                       const float* __restrict__ conv_b,
                       _Float16* __restrict__ xcs) {
  int gidx = blockIdx.x * 256 + threadIdx.x;
  int row = gidx >> 8, c = gidx & 255;
  int t = row & 31, base = row - t;
  float acc = conv_b[c];
#pragma unroll
  for (int kk = 0; kk < 4; ++kk) {
    int tt = t + kk - 3;
    if (tt >= 0)
      acc += conv_w[c * 4 + kk] * (float)xz[(size_t)(base + tt) * 512 + c];
  }
  acc = acc / (1.f + __expf(-acc)); // silu
  xcs[(size_t)row * 256 + c] = (_Float16)acc;
}

// ---------------- 2c: x_proj (256 -> 40) --------------------------------

__global__ void k_xproj(const _Float16* __restrict__ xcs,
                        const float* __restrict__ x_proj,
                        float* __restrict__ dbc) {
  int gidx = blockIdx.x * 256 + threadIdx.x;
  if (gidx >= ROWS * 40) return;
  int row = gidx / 40, j = gidx % 40;
  float acc = 0.f;
  for (int kk = 0; kk < 256; kk += 8) {
    v8h xv = *(const v8h*)(xcs + (size_t)row * 256 + kk);
#pragma unroll
    for (int jj = 0; jj < 8; ++jj)
      acc += (float)xv[jj] * x_proj[(kk + jj) * 40 + j];
  }
  dbc[(size_t)row * 40 + j] = acc;
}

// ---------------- 2d: dt = softplus(dbc[:,:8] @ dt_w + dt_b) ------------

__global__ void k_dt(const float* __restrict__ dbc,
                     const float* __restrict__ dt_w,
                     const float* __restrict__ dt_b, float* __restrict__ dt) {
  int gidx = blockIdx.x * 256 + threadIdx.x;
  int row = gidx >> 8, c = gidx & 255;
  float acc = dt_b[c];
#pragma unroll
  for (int kk = 0; kk < 8; ++kk)
    acc += dbc[(size_t)row * 40 + kk] * dt_w[kk * 256 + c];
  acc = (acc > 20.f) ? acc : log1pf(__expf(acc)); // softplus
  dt[(size_t)row * 256 + c] = acc;
}

// ---------------- 2e: selective scan (sequential in T) ------------------

__global__ void k_scan(const float* __restrict__ dt,
                       const _Float16* __restrict__ xcs,
                       const float* __restrict__ dbc,
                       const _Float16* __restrict__ xz,
                       const float* __restrict__ A_log,
                       const float* __restrict__ Dp, _Float16* __restrict__ y) {
  int gidx = blockIdx.x * 256 + threadIdx.x; // 512*256 threads
  int bn = gidx >> 8, c = gidx & 255;
  float As[16], h[16];
#pragma unroll
  for (int s = 0; s < 16; ++s) {
    As[s] = -__expf(A_log[c * 16 + s]);
    h[s] = 0.f;
  }
  float dpc = Dp[c];
  for (int t = 0; t < 32; ++t) {
    int row = bn * 32 + t;
    float dtv = dt[(size_t)row * 256 + c];
    float xv = (float)xcs[(size_t)row * 256 + c];
    float dx = dtv * xv;
    float yv = 0.f;
#pragma unroll
    for (int s = 0; s < 16; ++s) {
      float Bv = dbc[(size_t)row * 40 + 8 + s];
      float Cv = dbc[(size_t)row * 40 + 24 + s];
      h[s] = __expf(dtv * As[s]) * h[s] + dx * Bv;
      yv += h[s] * Cv;
    }
    yv += dpc * xv;
    float zv = (float)xz[(size_t)row * 512 + 256 + c];
    yv *= zv / (1.f + __expf(-zv)); // gate with silu(z)
    y[(size_t)row * 256 + c] = (_Float16)yv;
  }
}

// ---------------- 2f: out_w GEMM + residual + LayerNorm -> output -------

__global__ void k_out_ln(const _Float16* __restrict__ y,
                         const float* __restrict__ out_w,
                         const float* __restrict__ out_b,
                         const float* __restrict__ h2,
                         const float* __restrict__ g,
                         const float* __restrict__ bb, float* __restrict__ out) {
  __shared__ __align__(32) _Float16 sA[128 * 32];
  __shared__ __align__(32) _Float16 sBf[32 * 128];
  int tid = threadIdx.x, lane = tid & 31, wid = tid >> 5;
  int hf = (lane >> 4) & 1, nc = lane & 15;
  int rowbase = blockIdx.x * 128, mbase = wid * 16;

  v8f acc[8];
#pragma unroll
  for (int nt = 0; nt < 8; ++nt) acc[nt] = zero8();

  for (int kc = 0; kc < 8; ++kc) {
    __syncthreads();
    for (int i = tid * 8; i < 128 * 32; i += 256 * 8) {
      int rr = i >> 5, cc = i & 31;
      *(v8h*)(sA + i) =
          *(const v8h*)(y + (size_t)(rowbase + rr) * 256 + kc * 32 + cc);
    }
    stage_Bfrag_f32<32, 128>(sBf, out_w + (size_t)kc * 32 * 128, 128, 0, tid);
    __syncthreads();
    v16h a = load_A16(sA, 32, mbase, 0, lane);
#pragma unroll
    for (int nt = 0; nt < 8; ++nt) {
      v16h b = load_frag(sBf, nt, lane);
      acc[nt] = wmma16(a, b, acc[nt]);
    }
  }

  // epilogue: + out_b + residual(h2), LayerNorm, write (B,T,N,D) layout
#pragma unroll
  for (int nt = 0; nt < 8; ++nt)
#pragma unroll
    for (int r = 0; r < 8; ++r) {
      int r2 = rowbase + mbase + r + 8 * hf;       // (b,n,t) order
      int b = r2 >> 12, n = (r2 >> 5) & 127, t = r2 & 31;
      int rh = b * 4096 + t * 128 + n;             // (b,t,n) order
      int col = nt * 16 + nc;
      acc[nt][r] += out_b[col] + h2[(size_t)rh * 128 + col];
    }
#pragma unroll
  for (int r = 0; r < 8; ++r) {
    float s1 = 0.f, s2 = 0.f;
#pragma unroll
    for (int nt = 0; nt < 8; ++nt) {
      s1 += acc[nt][r];
      s2 += acc[nt][r] * acc[nt][r];
    }
    s1 = hsum16(s1);
    s2 = hsum16(s2);
    float mean = s1 * (1.f / 128.f);
    float var = s2 * (1.f / 128.f) - mean * mean;
    float rstd = rsqrtf(var + 1e-5f);
    int r2 = rowbase + mbase + r + 8 * hf;
    int b = r2 >> 12, n = (r2 >> 5) & 127, t = r2 & 31;
    int rh = b * 4096 + t * 128 + n;
#pragma unroll
    for (int nt = 0; nt < 8; ++nt) {
      int col = nt * 16 + nc;
      out[(size_t)rh * 128 + col] =
          (acc[nt][r] - mean) * rstd * g[col] + bb[col];
    }
  }
}

// ---------------- host driver -------------------------------------------

extern "C" void kernel_launch(void* const* d_in, const int* in_sizes, int n_in,
                              void* d_out, int out_size, void* d_ws,
                              size_t ws_size, hipStream_t stream) {
  const float* x       = (const float*)d_in[0];
  const float* adj     = (const float*)d_in[1];
  const float* Wq      = (const float*)d_in[2];
  const float* Wk      = (const float*)d_in[3];
  const float* Wv      = (const float*)d_in[4];
  const float* Wo      = (const float*)d_in[5];
  const float* bo      = (const float*)d_in[6];
  const float* ns_g    = (const float*)d_in[7];
  const float* ns_b    = (const float*)d_in[8];
  const float* nt_g    = (const float*)d_in[9];
  const float* nt_b    = (const float*)d_in[10];
  const float* in_proj = (const float*)d_in[11];
  const float* conv_w  = (const float*)d_in[12];
  const float* conv_b  = (const float*)d_in[13];
  const float* x_proj  = (const float*)d_in[14];
  const float* dt_w    = (const float*)d_in[15];
  const float* dt_b    = (const float*)d_in[16];
  const float* A_log   = (const float*)d_in[17];
  const float* Dp      = (const float*)d_in[18];
  const float* out_w   = (const float*)d_in[19];
  const float* out_b   = (const float*)d_in[20];
  float* out = (float*)d_out;

  char* w = (char*)d_ws;
  size_t off = 0;
  _Float16* qf  = (_Float16*)(w + off); off += (size_t)ROWS * 128 * 2;
  _Float16* kf  = (_Float16*)(w + off); off += (size_t)ROWS * 128 * 2;
  _Float16* vf  = (_Float16*)(w + off); off += (size_t)ROWS * 128 * 2;
  float*    of  = (float*)(w + off);    off += (size_t)ROWS * 128 * 4;
  float*    h2  = (float*)(w + off);    off += (size_t)ROWS * 128 * 4;
  _Float16* h2h = (_Float16*)(w + off); off += (size_t)ROWS * 128 * 2;
  _Float16* xz  = (_Float16*)(w + off); off += (size_t)ROWS * 512 * 2;
  _Float16* xcs = (_Float16*)(w + off); off += (size_t)ROWS * 256 * 2;
  float*    dbc = (float*)(w + off);    off += (size_t)ROWS * 40 * 4;
  float*    dtb = (float*)(w + off);    off += (size_t)ROWS * 256 * 4;
  _Float16* yb  = (_Float16*)(w + off); off += (size_t)ROWS * 256 * 2;

  k_qkv<<<dim3(128), dim3(256), 0, stream>>>(x, Wq, Wk, Wv, qf, kf, vf);
  k_attn<<<dim3(128, 4), dim3(256), 0, stream>>>(qf, kf, vf, adj, of);
  k_proj_ln<<<dim3(128), dim3(256), 0, stream>>>(of, Wo, bo, x, ns_g, ns_b, h2,
                                                 h2h);
  k_inproj<<<dim3(128, 4), dim3(256), 0, stream>>>(h2h, in_proj, xz);
  k_conv<<<dim3(ROWS), dim3(256), 0, stream>>>(xz, conv_w, conv_b, xcs);
  k_xproj<<<dim3((ROWS * 40 + 255) / 256), dim3(256), 0, stream>>>(xcs, x_proj,
                                                                   dbc);
  k_dt<<<dim3(ROWS), dim3(256), 0, stream>>>(dbc, dt_w, dt_b, dtb);
  k_scan<<<dim3(BN), dim3(256), 0, stream>>>(dtb, xcs, dbc, xz, A_log, Dp, yb);
  k_out_ln<<<dim3(128), dim3(256), 0, stream>>>(yb, out_w, out_b, h2, nt_g,
                                                nt_b, out);
}
